// NonstationarityTransformer_5909875000363
// MI455X (gfx1250) — matrix operand order repaired and checked by reading, not compiled, not run on hardware
//
#include <hip/hip_runtime.h>
#include <hip/hip_bf16.h>
#include <math.h>

typedef __attribute__((ext_vector_type(16))) _Float16 v16h;
typedef __attribute__((ext_vector_type(8)))  _Float16 v8h;
typedef __attribute__((ext_vector_type(8)))  float    v8f;

#define B_  32
#define L_  512
#define C_  64
#define D_  512
#define H_  8
#define DK_ 64
#define EL_ 3
#define DFF_ 2048
#define NC_ 8
#define PH_ 256
#define M_  (B_ * L_)   // 16384

#ifndef __has_builtin
#define __has_builtin(x) 0
#endif
#if defined(__HIP_DEVICE_COMPILE__)
#if __has_builtin(__builtin_amdgcn_global_load_async_to_lds_b128) && \
    __has_builtin(__builtin_amdgcn_s_wait_asynccnt)
#define USE_ASYNC 1
#endif
#endif
#ifndef USE_ASYNC
#define USE_ASYNC 0
#endif

// ---------------------------------------------------------------------------
// WMMA helpers (wave32, gfx1250 layouts per CDNA5 ISA 7.12.2)
// ---------------------------------------------------------------------------
__device__ __forceinline__ v16h make_v16(v8h lo, v8h hi) {
  v16h r;
#pragma unroll
  for (int i = 0; i < 8; ++i) { r[i] = lo[i]; r[i + 8] = hi[i]; }
  return r;
}

// A-matrix 16x32 f16 fragment from row-major [m][k] tile (ld halves per row).
__device__ __forceinline__ v16h lds_load_a(const _Float16* tile, int ld, int m,
                                           int k0, int hi) {
  const _Float16* p = tile + m * ld + k0 + 8 * hi;
  v8h lo = *(const v8h*)p;
  v8h hc = *(const v8h*)(p + 16);
  return make_v16(lo, hc);
}

// B-matrix 32x16 f16 fragment from an [n][k] (K-contiguous) tile.
__device__ __forceinline__ v16h lds_load_b(const _Float16* tileT, int ld, int n,
                                           int k0, int hi) {
  const _Float16* p = tileT + n * ld + k0 + 8 * hi;
  v8h lo = *(const v8h*)p;
  v8h hc = *(const v8h*)(p + 16);
  return make_v16(lo, hc);
}

__device__ __forceinline__ v8f wmma_f16(v16h a, v16h b, v8f c) {
  return __builtin_amdgcn_wmma_f32_16x16x32_f16(false, a, false, b,
                                                (short)0, c, false, false);
}

__device__ __forceinline__ float gelu_f(float v) {
  return 0.5f * v * (1.0f + erff(v * 0.70710678118654752f));
}

#if USE_ASYNC
typedef int v4i __attribute__((vector_size(16)));
typedef __attribute__((address_space(1))) v4i g_v4i;   // global
typedef __attribute__((address_space(3))) v4i l_v4i;   // LDS
__device__ __forceinline__ void async_cp16(_Float16* lds, const _Float16* g) {
  __builtin_amdgcn_global_load_async_to_lds_b128((g_v4i*)g, (l_v4i*)lds, 0, 0);
}
__device__ __forceinline__ void async_wait() {
  __builtin_amdgcn_s_wait_asynccnt(0);
}
#endif

// ---------------------------------------------------------------------------
// Pipelined WMMA GEMM:  Y = X[M,K] @ W[N,K]^T   (f16 in, f32 accumulate)
// block tile 128x64, KSTEP 64, double-buffered LDS, 256 thr = 8 waves.
// mode 0: Y[f32] = acc
// mode 1: Yh[f16] = acc + bias[n]
// mode 2: Yh[f16] = gelu(acc + bias[n])
// requires: M%128==0, N%64==0, K%64==0
// ---------------------------------------------------------------------------
__global__ void gemm_wmma(const _Float16* __restrict__ X,
                          const _Float16* __restrict__ W,
                          float* __restrict__ Y,
                          _Float16* __restrict__ Yh,
                          const float* __restrict__ bias,
                          int M, int N, int K, int mode) {
  __shared__ _Float16 As[2][128 * 64];
  __shared__ _Float16 Bs[2][64 * 64];
  const int tid  = threadIdx.x;
  const int wave = tid >> 5, lane = tid & 31;
  const int hi = lane >> 4, ln = lane & 15;
  const int m0 = blockIdx.x * 128, n0 = blockIdx.y * 64;

  // per-thread staging coordinates (fixed) + running global source pointers
  int aoff[4], boff[2];
  const _Float16* xsrc[4];
  const _Float16* wsrc[2];
#pragma unroll
  for (int t = 0; t < 4; ++t) {
    int i = tid + t * 256, r = i >> 3, cc = (i & 7) * 8;
    aoff[t] = r * 64 + cc;
    xsrc[t] = &X[(size_t)(m0 + r) * K + cc];
  }
#pragma unroll
  for (int t = 0; t < 2; ++t) {
    int i = tid + t * 256, r = i >> 3, cc = (i & 7) * 8;
    boff[t] = r * 64 + cc;
    wsrc[t] = &W[(size_t)(n0 + r) * K + cc];
  }

  // ---- prologue: fill buffer 0 with k-slice [0,64) ------------------------
#if USE_ASYNC
#pragma unroll
  for (int t = 0; t < 4; ++t) async_cp16(&As[0][aoff[t]], xsrc[t]);
#pragma unroll
  for (int t = 0; t < 2; ++t) async_cp16(&Bs[0][boff[t]], wsrc[t]);
  async_wait();
#else
#pragma unroll
  for (int t = 0; t < 4; ++t) *(v8h*)&As[0][aoff[t]] = *(const v8h*)xsrc[t];
#pragma unroll
  for (int t = 0; t < 2; ++t) *(v8h*)&Bs[0][boff[t]] = *(const v8h*)wsrc[t];
#endif
#pragma unroll
  for (int t = 0; t < 4; ++t) xsrc[t] += 64;
#pragma unroll
  for (int t = 0; t < 2; ++t) wsrc[t] += 64;
  __syncthreads();

  v8f acc[4] = {};
  int p = 0;
  for (int k0 = 0; k0 < K; k0 += 64) {
    const bool pf = (k0 + 64) < K;
#if USE_ASYNC
    if (pf) {
      _Float16* Ad = As[1 - p];
      _Float16* Bd = Bs[1 - p];
#pragma unroll
      for (int t = 0; t < 4; ++t) async_cp16(Ad + aoff[t], xsrc[t]);
#pragma unroll
      for (int t = 0; t < 2; ++t) async_cp16(Bd + boff[t], wsrc[t]);
    }
#else
    v8h areg[4], breg[2];
    if (pf) {
#pragma unroll
      for (int t = 0; t < 4; ++t) areg[t] = *(const v8h*)xsrc[t];
#pragma unroll
      for (int t = 0; t < 2; ++t) breg[t] = *(const v8h*)wsrc[t];
    }
#endif
#pragma unroll
    for (int t = 0; t < 4; ++t) xsrc[t] += 64;
#pragma unroll
    for (int t = 0; t < 2; ++t) wsrc[t] += 64;

    // ---- batch fragment loads, then 8 WMMAs on current buffer -------------
    const _Float16* Ap = As[p] + (wave << 4) * 64;
    const _Float16* Bp = Bs[p];
    v16h a0 = lds_load_a(Ap, 64, ln, 0, hi);
    v16h a1 = lds_load_a(Ap, 64, ln, 32, hi);
    v16h bf[8];
#pragma unroll
    for (int j = 0; j < 4; ++j) {
      bf[2 * j]     = lds_load_b(Bp + (j * 16) * 64, 64, ln, 0, hi);
      bf[2 * j + 1] = lds_load_b(Bp + (j * 16) * 64, 64, ln, 32, hi);
    }
#pragma unroll
    for (int j = 0; j < 4; ++j) {
      acc[j] = wmma_f16(a0, bf[2 * j],     acc[j]);
      acc[j] = wmma_f16(a1, bf[2 * j + 1], acc[j]);
    }
#if USE_ASYNC
    async_wait();
#else
    if (pf) {
      _Float16* Ad = As[1 - p];
      _Float16* Bd = Bs[1 - p];
#pragma unroll
      for (int t = 0; t < 4; ++t) *(v8h*)&Ad[aoff[t]] = areg[t];
#pragma unroll
      for (int t = 0; t < 2; ++t) *(v8h*)&Bd[boff[t]] = breg[t];
    }
#endif
    __syncthreads();
    p ^= 1;
  }

  // ---- fused epilogue -----------------------------------------------------
  const int mbase = m0 + (wave << 4) + 8 * hi;
  if (mode == 0) {
#pragma unroll
    for (int j = 0; j < 4; ++j)
#pragma unroll
      for (int r = 0; r < 8; ++r)
        Y[(size_t)(mbase + r) * N + n0 + j * 16 + ln] = acc[j][r];
  } else {
#pragma unroll
    for (int j = 0; j < 4; ++j) {
      float bb = bias[n0 + j * 16 + ln];
#pragma unroll
      for (int r = 0; r < 8; ++r) {
        float v = acc[j][r] + bb;
        if (mode == 2) v = gelu_f(v);
        Yh[(size_t)(mbase + r) * N + n0 + j * 16 + ln] = (_Float16)v;
      }
    }
  }
}

// ---------------------------------------------------------------------------
// Pipelined flash attention with WMMA. grid (L/128, H, B), 256 thr = 8 waves.
// logits = ts*(q.k) + sdelta[s], ts = tau[b]/sqrt(DK), sdelta = delta*scale
// ---------------------------------------------------------------------------
__global__ void attn_wmma(const _Float16* __restrict__ Qh,
                          const _Float16* __restrict__ Kh,
                          const _Float16* __restrict__ Vh,
                          const float* __restrict__ tau,
                          const float* __restrict__ delta,
                          _Float16* __restrict__ Oh) {
  __shared__ _Float16 Qs[128 * 64];
  __shared__ _Float16 Ks[2][32 * 64];
  __shared__ _Float16 Vt[2][64 * 32];   // transposed: [d][s]
  __shared__ _Float16 Ps[8][16 * 32];
  __shared__ float    Ds[L_];

  const int q0 = blockIdx.x * 128;
  const int h  = blockIdx.y;
  const int b  = blockIdx.z;
  const int tid  = threadIdx.x;
  const int wave = tid >> 5, lane = tid & 31;
  const int hi = lane >> 4, ln = lane & 15;

  const float scale = 0.125f;             // 1/sqrt(64)
  const float ts = tau[b] * scale;

  const _Float16* Qg = Qh + ((size_t)b * L_) * D_ + h * DK_;
  const _Float16* Kg = Kh + ((size_t)b * L_) * D_ + h * DK_;
  const _Float16* Vg = Vh + ((size_t)b * L_) * D_ + h * DK_;

  // per-thread staging coords + running source pointers (advance 32 rows)
  const int kr = tid >> 3, kc = (tid & 7) * 8;          // K tile: 32x64 halves
  const _Float16* ksrc = &Kg[(size_t)kr * D_ + kc];
  int vd[8], vs[8];
  const _Float16* vsrc[8];
#pragma unroll
  for (int t = 0; t < 8; ++t) {
    int i = tid + t * 256;
    vd[t] = i & 63; vs[t] = i >> 6;
    vsrc[t] = &Vg[(size_t)vs[t] * D_ + vd[t]];
  }

  // ---- prologue: Ds, Q tile, K/V tile 0 -----------------------------------
  for (int i = tid; i < L_; i += 256) Ds[i] = delta[b * L_ + i] * scale;
#pragma unroll
  for (int t = 0; t < 4; ++t) {
    int i = tid + t * 256, r = i >> 3, c = (i & 7) * 8;
    *(v8h*)&Qs[r * 64 + c] = *(const v8h*)&Qg[(size_t)(q0 + r) * D_ + c];
  }
  *(v8h*)&Ks[0][kr * 64 + kc] = *(const v8h*)ksrc;
#pragma unroll
  for (int t = 0; t < 8; ++t) Vt[0][vd[t] * 32 + vs[t]] = *vsrc[t];
  ksrc += (size_t)32 * D_;
#pragma unroll
  for (int t = 0; t < 8; ++t) vsrc[t] += (size_t)32 * D_;
  __syncthreads();

  const int mrow = wave << 4;
  v16h qa0 = lds_load_a(Qs + mrow * 64, 64, ln, 0, hi);
  v16h qa1 = lds_load_a(Qs + mrow * 64, 64, ln, 32, hi);

  v8f acc[4] = {};
  float mrun[8], lrun[8];
#pragma unroll
  for (int r = 0; r < 8; ++r) { mrun[r] = -1e30f; lrun[r] = 0.0f; }

  int p = 0;
  for (int s0 = 0; s0 < L_; s0 += 32) {
    const bool pf = (s0 + 32) < L_;
    v8h kreg;
    _Float16 vreg[8];
    if (pf) {                                    // prefetch next K/V to regs
      kreg = *(const v8h*)ksrc;
#pragma unroll
      for (int t = 0; t < 8; ++t) vreg[t] = *vsrc[t];
    }
    ksrc += (size_t)32 * D_;
#pragma unroll
    for (int t = 0; t < 8; ++t) vsrc[t] += (size_t)32 * D_;

    const _Float16* Ksp = Ks[p];
    const _Float16* Vtp = Vt[p];

    // ---- scores: batch 4 fragment loads, then 4 WMMAs ---------------------
    v16h kb0 = lds_load_b(Ksp, 64, ln, 0, hi);
    v16h kb1 = lds_load_b(Ksp, 64, ln, 32, hi);
    v16h kb2 = lds_load_b(Ksp + 16 * 64, 64, ln, 0, hi);
    v16h kb3 = lds_load_b(Ksp + 16 * 64, 64, ln, 32, hi);
    v8f sc0 = {}, sc1 = {};
    sc0 = wmma_f16(qa0, kb0, sc0);
    sc0 = wmma_f16(qa1, kb1, sc0);
    sc1 = wmma_f16(qa0, kb2, sc1);
    sc1 = wmma_f16(qa1, kb3, sc1);

    const float d0 = Ds[s0 + ln];
    const float d1 = Ds[s0 + 16 + ln];

    float p0[8], p1[8], fcor[8];
#pragma unroll
    for (int r = 0; r < 8; ++r) {
      float a = ts * sc0[r] + d0;
      float c = ts * sc1[r] + d1;
      float mx = fmaxf(a, c);
#pragma unroll
      for (int off = 1; off < 16; off <<= 1)
        mx = fmaxf(mx, __shfl_xor(mx, off, 32));
      float mnew = fmaxf(mrun[r], mx);
      float f  = __expf(mrun[r] - mnew);
      float e0 = __expf(a - mnew);
      float e1 = __expf(c - mnew);
      float rs = e0 + e1;
#pragma unroll
      for (int off = 1; off < 16; off <<= 1)
        rs += __shfl_xor(rs, off, 32);
      lrun[r] = lrun[r] * f + rs;
      mrun[r] = mnew;
      fcor[r] = f; p0[r] = e0; p1[r] = e1;
    }
#pragma unroll
    for (int j = 0; j < 4; ++j)
#pragma unroll
      for (int r = 0; r < 8; ++r) acc[j][r] *= fcor[r];

    // ---- stage P through per-wave LDS to become an A fragment -------------
    _Float16* P = Ps[wave];
#pragma unroll
    for (int r = 0; r < 8; ++r) {
      P[(r + 8 * hi) * 32 + ln]      = (_Float16)p0[r];
      P[(r + 8 * hi) * 32 + 16 + ln] = (_Float16)p1[r];
    }
    v16h pa = lds_load_a(P, 32, ln, 0, hi);   // DS in-order per wave
    v16h vb0 = lds_load_b(Vtp,             32, ln, 0, hi);
    v16h vb1 = lds_load_b(Vtp + 16 * 32,   32, ln, 0, hi);
    v16h vb2 = lds_load_b(Vtp + 32 * 32,   32, ln, 0, hi);
    v16h vb3 = lds_load_b(Vtp + 48 * 32,   32, ln, 0, hi);
    acc[0] = wmma_f16(pa, vb0, acc[0]);
    acc[1] = wmma_f16(pa, vb1, acc[1]);
    acc[2] = wmma_f16(pa, vb2, acc[2]);
    acc[3] = wmma_f16(pa, vb3, acc[3]);

    if (pf) {                                  // commit prefetched K/V tile
      *(v8h*)&Ks[1 - p][kr * 64 + kc] = kreg;
#pragma unroll
      for (int t = 0; t < 8; ++t) Vt[1 - p][vd[t] * 32 + vs[t]] = vreg[t];
    }
    __syncthreads();
    p ^= 1;
  }

  _Float16* Og = Oh + ((size_t)b * L_) * D_ + h * DK_;
  const int mg = q0 + mrow + 8 * hi;
  float rinv[8];
#pragma unroll
  for (int r = 0; r < 8; ++r) rinv[r] = 1.0f / lrun[r];
#pragma unroll
  for (int j = 0; j < 4; ++j)
#pragma unroll
    for (int r = 0; r < 8; ++r)
      Og[(size_t)(mg + r) * D_ + j * 16 + ln] = (_Float16)(acc[j][r] * rinv[r]);
}

// ---------------------------------------------------------------------------
// Small / elementwise kernels
// ---------------------------------------------------------------------------
__global__ void meanstd_kernel(const float* __restrict__ xe,
                               float* __restrict__ mv, float* __restrict__ sv) {
  const int bc = blockIdx.x;                 // b*64 + c
  const int b = bc >> 6, c = bc & 63;
  const int tid = threadIdx.x;
  __shared__ float s1[256], s2[256];
  float s = 0.f, q = 0.f;
  for (int l = tid; l < L_; l += 256) {
    float v = xe[((size_t)b * L_ + l) * C_ + c];
    s += v; q += v * v;
  }
  s1[tid] = s; s2[tid] = q; __syncthreads();
  for (int st = 128; st > 0; st >>= 1) {
    if (tid < st) { s1[tid] += s1[tid + st]; s2[tid] += s2[tid + st]; }
    __syncthreads();
  }
  if (tid == 0) {
    float m = s1[0] * (1.0f / L_);
    mv[bc] = m;
    sv[bc] = sqrtf(s2[0] * (1.0f / L_) - m * m + 1e-5f);
  }
}

__global__ void projector_kernel(const float* __restrict__ xe,
                                 const float* __restrict__ stats,   // (B,C)
                                 const float* __restrict__ cw,      // (L,3)
                                 const float* __restrict__ w1, const float* __restrict__ b1,
                                 const float* __restrict__ w2, const float* __restrict__ b2,
                                 const float* __restrict__ w3, int w3_rows,
                                 float* __restrict__ out, int apply_exp) {
  const int b = blockIdx.x;
  const int tid = threadIdx.x;
  __shared__ float h0[2 * C_], h1[PH_], h2[PH_], red[256];

  const int c = tid >> 2, sub = tid & 3;
  const int cm1 = (c + C_ - 1) & (C_ - 1), cp1 = (c + 1) & (C_ - 1);
  float acc = 0.f;
  for (int s = sub; s < L_; s += 4) {
    const float* xr = xe + ((size_t)b * L_ + s) * C_;
    const float* cr = cw + s * 3;
    acc += xr[cm1] * cr[0] + xr[c] * cr[1] + xr[cp1] * cr[2];
  }
  red[tid] = acc; __syncthreads();
  if (sub == 0) h0[c] = red[tid] + red[tid + 1] + red[tid + 2] + red[tid + 3];
  if (tid < C_) h0[C_ + tid] = stats[b * C_ + tid];
  __syncthreads();

  { float a = b1[tid]; const float* wr = w1 + tid * (2 * C_);
    for (int i = 0; i < 2 * C_; ++i) a += h0[i] * wr[i];
    h1[tid] = fmaxf(a, 0.f); }
  __syncthreads();
  { float a = b2[tid]; const float* wr = w2 + tid * PH_;
    for (int i = 0; i < PH_; ++i) a += h1[i] * wr[i];
    h2[tid] = fmaxf(a, 0.f); }
  __syncthreads();
  for (int rI = tid; rI < w3_rows; rI += 256) {
    const float* wr = w3 + rI * PH_;
    float a = 0.f;
    for (int i = 0; i < PH_; ++i) a += h2[i] * wr[i];
    out[b * w3_rows + rI] = apply_exp ? __expf(a) : a;
  }
}

__global__ void pack_win_kernel(const float* __restrict__ xe,
                                _Float16* __restrict__ wout, int n) {
  int i = blockIdx.x * 256 + threadIdx.x;
  if (i >= n) return;
  int ck = i % 192;
  int l  = (i / 192) & (L_ - 1);
  int b  = i / (192 * L_);
  int c = ck / 3, k = ck - 3 * c;
  int ls = (l - 1 + k + L_) & (L_ - 1);
  wout[i] = (_Float16)xe[((size_t)b * L_ + ls) * C_ + c];
}

__global__ void cvt_f16_kernel(const float* __restrict__ s,
                               _Float16* __restrict__ d, int n) {
  int i = blockIdx.x * 256 + threadIdx.x;
  if (i < n) d[i] = (_Float16)s[i];
}

__global__ void peadd_cvt_kernel(const float* __restrict__ t,
                                 float* __restrict__ x,
                                 _Float16* __restrict__ xh, int n) {
  int i = blockIdx.x * 256 + threadIdx.x;
  if (i >= n) return;
  int d = i & (D_ - 1);
  int l = (i >> 9) & (L_ - 1);
  int pair = d >> 1;
  float freq = __expf((float)(2 * pair) * (-9.210340371976184f / (float)D_));
  float ang = (float)l * freq;
  float pe = (d & 1) ? __cosf(ang) : __sinf(ang);
  float v = t[i] + pe;
  x[i] = v;
  xh[i] = (_Float16)v;
}

// x = LN(x + G + bias; gamma, beta); also emit f16 copy. One block per row.
__global__ void bias_res_ln_kernel(const float* __restrict__ G,
                                   const float* __restrict__ bias,
                                   const float* __restrict__ gam,
                                   const float* __restrict__ bet,
                                   float* __restrict__ x,
                                   _Float16* __restrict__ xh) {
  const int row = blockIdx.x, tid = threadIdx.x;
  __shared__ float sh[256];
  const size_t base = (size_t)row * D_;
  float v0 = x[base + tid]       + G[base + tid]       + bias[tid];
  float v1 = x[base + tid + 256] + G[base + tid + 256] + bias[tid + 256];
  sh[tid] = v0 + v1; __syncthreads();
  for (int st = 128; st > 0; st >>= 1) { if (tid < st) sh[tid] += sh[tid + st]; __syncthreads(); }
  float mean = sh[0] * (1.0f / D_);
  __syncthreads();
  float d0 = v0 - mean, d1 = v1 - mean;
  sh[tid] = d0 * d0 + d1 * d1; __syncthreads();
  for (int st = 128; st > 0; st >>= 1) { if (tid < st) sh[tid] += sh[tid + st]; __syncthreads(); }
  float inv = rsqrtf(sh[0] * (1.0f / D_) + 1e-5f);
  float o0 = d0 * inv * gam[tid] + bet[tid];
  float o1 = d1 * inv * gam[tid + 256] + bet[tid + 256];
  x[base + tid] = o0;            x[base + tid + 256] = o1;
  xh[base + tid] = (_Float16)o0; xh[base + tid + 256] = (_Float16)o1;
}

// xg = gelu(LN(x; g, b))
__global__ void final_norm_gelu_kernel(const float* __restrict__ x,
                                       const float* __restrict__ gam,
                                       const float* __restrict__ bet,
                                       float* __restrict__ xg) {
  const int row = blockIdx.x, tid = threadIdx.x;
  __shared__ float sh[256];
  const size_t base = (size_t)row * D_;
  float v0 = x[base + tid], v1 = x[base + tid + 256];
  sh[tid] = v0 + v1; __syncthreads();
  for (int st = 128; st > 0; st >>= 1) { if (tid < st) sh[tid] += sh[tid + st]; __syncthreads(); }
  float mean = sh[0] * (1.0f / D_);
  __syncthreads();
  float d0 = v0 - mean, d1 = v1 - mean;
  sh[tid] = d0 * d0 + d1 * d1; __syncthreads();
  for (int st = 128; st > 0; st >>= 1) { if (tid < st) sh[tid] += sh[tid + st]; __syncthreads(); }
  float inv = rsqrtf(sh[0] * (1.0f / D_) + 1e-5f);
  float o0 = d0 * inv * gam[tid] + bet[tid];
  float o1 = d1 * inv * gam[tid + 256] + bet[tid + 256];
  xg[base + tid]       = gelu_f(o0);
  xg[base + tid + 256] = gelu_f(o1);
}

// out[b,nc] = dot(xg[b,:], proj_w[nc,:]) + proj_b[nc];  grid (B, NC)
__global__ void final_proj_kernel(const float* __restrict__ xg,
                                  const float* __restrict__ pw,
                                  const float* __restrict__ pb,
                                  float* __restrict__ out) {
  const int b = blockIdx.x, nc = blockIdx.y, tid = threadIdx.x;
  __shared__ float sh[256];
  const float* xr = xg + (size_t)b * (D_ * L_);
  const float* wr = pw + (size_t)nc * (D_ * L_);
  float s = 0.f;
  for (int i = tid; i < D_ * L_; i += 256) s += xr[i] * wr[i];
  sh[tid] = s; __syncthreads();
  for (int st = 128; st > 0; st >>= 1) { if (tid < st) sh[tid] += sh[tid + st]; __syncthreads(); }
  if (tid == 0) out[b * NC_ + nc] = sh[0] + pb[nc];
}

// ---------------------------------------------------------------------------
// Launch
// ---------------------------------------------------------------------------
extern "C" void kernel_launch(void* const* d_in, const int* in_sizes, int n_in,
                              void* d_out, int out_size, void* d_ws, size_t ws_size,
                              hipStream_t stream) {
  (void)in_sizes; (void)n_in; (void)out_size; (void)ws_size;

  const float* x_enc      = (const float*)d_in[0];
  const float* emb_conv_w = (const float*)d_in[1];
  const float* tau_conv_w = (const float*)d_in[2];
  const float* tau_w1 = (const float*)d_in[3];
  const float* tau_b1 = (const float*)d_in[4];
  const float* tau_w2 = (const float*)d_in[5];
  const float* tau_b2 = (const float*)d_in[6];
  const float* tau_w3 = (const float*)d_in[7];
  const float* delta_conv_w = (const float*)d_in[8];
  const float* delta_w1 = (const float*)d_in[9];
  const float* delta_b1 = (const float*)d_in[10];
  const float* delta_w2 = (const float*)d_in[11];
  const float* delta_b2 = (const float*)d_in[12];
  const float* delta_w3 = (const float*)d_in[13];
  const float* Wq = (const float*)d_in[14];
  const float* bq = (const float*)d_in[15];
  const float* Wk = (const float*)d_in[16];
  const float* bk = (const float*)d_in[17];
  const float* Wv = (const float*)d_in[18];
  const float* bv = (const float*)d_in[19];
  const float* Wo = (const float*)d_in[20];
  const float* bo = (const float*)d_in[21];
  const float* ff_w1 = (const float*)d_in[22];
  const float* ff_b1 = (const float*)d_in[23];
  const float* ff_w2 = (const float*)d_in[24];
  const float* ff_b2 = (const float*)d_in[25];
  const float* ln1_g = (const float*)d_in[26];
  const float* ln1_b = (const float*)d_in[27];
  const float* ln2_g = (const float*)d_in[28];
  const float* ln2_b = (const float*)d_in[29];
  const float* norm_g = (const float*)d_in[30];
  const float* norm_b = (const float*)d_in[31];
  const float* proj_w = (const float*)d_in[32];
  const float* proj_b = (const float*)d_in[33];
  float* out = (float*)d_out;

  char* ws = (char*)d_ws;
  size_t off = 0;
  auto alloc = [&](size_t bytes) -> void* {
    void* p = ws + off;
    off += (bytes + 255) & ~(size_t)255;
    return p;
  };
  const size_t NBLD = (size_t)M_ * D_;           // 8,388,608
  const size_t NBLF = (size_t)M_ * DFF_;         // 33,554,432

  float*    x    = (float*)alloc(NBLD * 4);
  _Float16* xh   = (_Float16*)alloc(NBLD * 2);
  float*    tmp  = (float*)alloc(NBLF * 4);      // also reused as xg at the end
  _Float16* hh   = (_Float16*)alloc(NBLF * 2);
  _Float16* qh   = (_Float16*)alloc(NBLD * 2);
  _Float16* kh   = (_Float16*)alloc(NBLD * 2);
  _Float16* vh   = (_Float16*)alloc(NBLD * 2);
  _Float16* oh   = (_Float16*)alloc(NBLD * 2);
  _Float16* winh = (_Float16*)alloc((size_t)M_ * 192 * 2);
  float*    meanv = (float*)alloc((size_t)B_ * C_ * 4);
  float*    stdv  = (float*)alloc((size_t)B_ * C_ * 4);
  float*    tauv  = (float*)alloc((size_t)B_ * 4);
  float*    deltav = (float*)alloc((size_t)B_ * L_ * 4);
  _Float16* wqh = (_Float16*)alloc((size_t)EL_ * D_ * D_ * 2);
  _Float16* wkh = (_Float16*)alloc((size_t)EL_ * D_ * D_ * 2);
  _Float16* wvh = (_Float16*)alloc((size_t)EL_ * D_ * D_ * 2);
  _Float16* woh = (_Float16*)alloc((size_t)EL_ * D_ * D_ * 2);
  _Float16* w1h = (_Float16*)alloc((size_t)EL_ * DFF_ * D_ * 2);
  _Float16* w2h = (_Float16*)alloc((size_t)EL_ * D_ * DFF_ * 2);
  _Float16* wembh = (_Float16*)alloc((size_t)D_ * 192 * 2);
  float*    xg = tmp;

  // ---- stats + projectors -------------------------------------------------
  meanstd_kernel<<<B_ * C_, 256, 0, stream>>>(x_enc, meanv, stdv);
  projector_kernel<<<B_, 256, 0, stream>>>(x_enc, stdv, tau_conv_w,
      tau_w1, tau_b1, tau_w2, tau_b2, tau_w3, 1, tauv, 1);
  projector_kernel<<<B_, 256, 0, stream>>>(x_enc, meanv, delta_conv_w,
      delta_w1, delta_b1, delta_w2, delta_b2, delta_w3, L_, deltav, 0);

  // ---- weight conversion to f16 ------------------------------------------
  auto cvt = [&](const float* s, _Float16* d, int n) {
    cvt_f16_kernel<<<(n + 255) / 256, 256, 0, stream>>>(s, d, n);
  };
  cvt(Wq, wqh, EL_ * D_ * D_);
  cvt(Wk, wkh, EL_ * D_ * D_);
  cvt(Wv, wvh, EL_ * D_ * D_);
  cvt(Wo, woh, EL_ * D_ * D_);
  cvt(ff_w1, w1h, EL_ * DFF_ * D_);
  cvt(ff_w2, w2h, EL_ * D_ * DFF_);
  cvt(emb_conv_w, wembh, D_ * 192);

  // ---- embedding conv (WMMA GEMM over 192-wide windows) + PE --------------
  {
    int n = M_ * 192;
    pack_win_kernel<<<(n + 255) / 256, 256, 0, stream>>>(x_enc, winh, n);
    gemm_wmma<<<dim3(M_ / 128, D_ / 64), 256, 0, stream>>>(
        winh, wembh, tmp, nullptr, nullptr, M_, D_, 192, 0);
    int ne = (int)NBLD;
    peadd_cvt_kernel<<<(ne + 255) / 256, 256, 0, stream>>>(tmp, x, xh, ne);
  }

  // ---- encoder layers -----------------------------------------------------
  for (int i = 0; i < EL_; ++i) {
    const size_t wOff = (size_t)i * D_ * D_;
    const size_t fOff = (size_t)i * DFF_ * D_;

    gemm_wmma<<<dim3(M_ / 128, D_ / 64), 256, 0, stream>>>(
        xh, wqh + wOff, tmp, qh, bq + i * D_, M_, D_, D_, 1);
    gemm_wmma<<<dim3(M_ / 128, D_ / 64), 256, 0, stream>>>(
        xh, wkh + wOff, tmp, kh, bk + i * D_, M_, D_, D_, 1);
    gemm_wmma<<<dim3(M_ / 128, D_ / 64), 256, 0, stream>>>(
        xh, wvh + wOff, tmp, vh, bv + i * D_, M_, D_, D_, 1);

    attn_wmma<<<dim3(L_ / 128, H_, B_), 256, 0, stream>>>(qh, kh, vh, tauv, deltav, oh);

    gemm_wmma<<<dim3(M_ / 128, D_ / 64), 256, 0, stream>>>(
        oh, woh + wOff, tmp, nullptr, nullptr, M_, D_, D_, 0);
    bias_res_ln_kernel<<<M_, 256, 0, stream>>>(tmp, bo + i * D_,
        ln1_g + i * D_, ln1_b + i * D_, x, xh);

    gemm_wmma<<<dim3(M_ / 128, DFF_ / 64), 256, 0, stream>>>(
        xh, w1h + fOff, tmp, hh, ff_b1 + i * DFF_, M_, DFF_, D_, 2);
    gemm_wmma<<<dim3(M_ / 128, D_ / 64), 256, 0, stream>>>(
        hh, w2h + fOff, tmp, nullptr, nullptr, M_, D_, DFF_, 0);
    bias_res_ln_kernel<<<M_, 256, 0, stream>>>(tmp, ff_b2 + i * D_,
        ln2_g + i * D_, ln2_b + i * D_, x, xh);
  }

  // ---- final norm + gelu + projection ------------------------------------
  final_norm_gelu_kernel<<<M_, 256, 0, stream>>>(x, norm_g, norm_b, xg);
  final_proj_kernel<<<dim3(B_, NC_), 256, 0, stream>>>(xg, proj_w, proj_b, out);
}
